// Performer_28879360098992
// MI455X (gfx1250) — matrix operand order repaired and checked by reading, not compiled, not run on hardware
//
#include <hip/hip_runtime.h>

typedef __bf16 bf16;
typedef __attribute__((ext_vector_type(16))) __bf16 v16bf;
typedef __attribute__((ext_vector_type(8)))  __bf16 v8bf;
typedef __attribute__((ext_vector_type(8)))  float  v8f;

#define LB __launch_bounds__(256)

constexpr int cB = 2, cL = 4096, cDM = 1024, cH = 8, cM = 256, cD = 128;

// ---------------------------------------------------------------------------
// Generic bf16 WMMA GEMM: C(z,i,j) = sum_k A(z,i,k) * B(z,k,j), fp32 accum.
// A element (i,k): A + (z&7)*a_z1 + (z>>3)*a_z8 + i*a_rs + k      (unit k-stride)
// B element (k,j): Bm + (z&7)*b_z1 + (z>>3)*b_z8 + k*b_rs + j     (unit n-stride)
// Block tile 128x128, BK=32, 8 waves (2x4), each wave 64x32 = 4x2 WMMA tiles.
// Double-buffered LDS staging; global loads for tile t+1 issued before the
// WMMAs of tile t; prefetch of tile t+2 via global_prefetch_b8.
// ---------------------------------------------------------------------------
struct GemmP {
    const bf16* A;  long a_z1, a_z8, a_rs;
    const bf16* Bm; long b_z1, b_z8, b_rs;
    float* outf; long of_ld;
    bf16*  outb; long ob_z1, ob_z8, ob_ld;
    const float* bias;
    const float* aux; long aux_ld;
    int K;
};

template <int EPI>
__global__ LB void gemm_bf16(GemmP p) {
    __shared__ __attribute__((aligned(16))) bf16 As[2][128 * 48];
    __shared__ __attribute__((aligned(16))) bf16 Bs[2][128 * 48];

    const int tid  = threadIdx.x;
    const int lane = tid & 31;
    const int wvi  = tid >> 5;     // 0..7
    const int wm   = wvi >> 2;     // 0..1  (64 rows each)
    const int wn   = wvi & 3;      // 0..3  (32 cols each)
    const int hf   = lane >> 4;    // half-wave
    const int l16  = lane & 15;
    const int z    = blockIdx.z;
    const long rowBase = (long)blockIdx.y * 128;
    const long colBase = (long)blockIdx.x * 128;

    const bf16* Ab = p.A  + (long)(z & 7) * p.a_z1 + (long)(z >> 3) * p.a_z8 + rowBase * p.a_rs;
    const bf16* Bb = p.Bm + (long)(z & 7) * p.b_z1 + (long)(z >> 3) * p.b_z8 + colBase;

    v8f acc[4][2];
    const v8f vzero = {0.f, 0.f, 0.f, 0.f, 0.f, 0.f, 0.f, 0.f};
#pragma unroll
    for (int i = 0; i < 4; i++)
#pragma unroll
        for (int j = 0; j < 2; j++) acc[i][j] = vzero;

    const int ar  = tid >> 1;         // A tile row 0..127
    const int ac  = (tid & 1) * 16;   // A tile col group (0/16)
    const int bkr = tid >> 3;         // B tile k-row 0..31
    const int bn0 = (tid & 7) * 16;   // B tile n group

    union U8 { v8bf v; bf16 e[8]; };

    // ---- prologue: stage tile 0 into buffer 0 ----
    {
        const bf16* sA = Ab + (long)ar * p.a_rs + ac;
        *(v8bf*)&As[0][ar * 48 + ac]     = *(const v8bf*)(sA);
        *(v8bf*)&As[0][ar * 48 + ac + 8] = *(const v8bf*)(sA + 8);
        const bf16* sB = Bb + (long)bkr * p.b_rs + bn0;
        U8 u0, u1;
        u0.v = *(const v8bf*)(sB);
        u1.v = *(const v8bf*)(sB + 8);
#pragma unroll
        for (int i = 0; i < 8; i++) Bs[0][(bn0 + i) * 48 + bkr] = u0.e[i];
#pragma unroll
        for (int i = 0; i < 8; i++) Bs[0][(bn0 + 8 + i) * 48 + bkr] = u1.e[i];
    }
    __syncthreads();

    for (int kt = 0; kt < p.K; kt += 32) {
        const int cur = (kt >> 5) & 1;
        const int nxt = cur ^ 1;
        const bool more = (kt + 32) < p.K;

        // 1) issue next tile's global loads (latency overlapped with WMMAs)
        v8bf na0 = {}, na1 = {};
        U8 nb0{}, nb1{};
        if (more) {
            const bf16* sA = Ab + (long)ar * p.a_rs + (kt + 32) + ac;
            na0 = *(const v8bf*)(sA);
            na1 = *(const v8bf*)(sA + 8);
            const bf16* sB = Bb + (long)(kt + 32 + bkr) * p.b_rs + bn0;
            nb0.v = *(const v8bf*)(sB);
            nb1.v = *(const v8bf*)(sB + 8);
            if (kt + 64 < p.K) {
                __builtin_prefetch(Ab + (long)ar * p.a_rs + (kt + 64) + ac, 0, 1);
                __builtin_prefetch(Bb + (long)(kt + 64 + bkr) * p.b_rs + bn0, 0, 1);
            }
        }

        // 2) compute from current buffer (fragments per ISA 16-bit layouts)
        union Frag { v16bf v; v8bf h[2]; };
        Frag fa[4], fb[2];
#pragma unroll
        for (int mi = 0; mi < 4; mi++) {
            const int r = wm * 64 + mi * 16 + l16;
            fa[mi].h[0] = *(const v8bf*)&As[cur][r * 48 + 8 * hf];       // K 0..7 / 8..15
            fa[mi].h[1] = *(const v8bf*)&As[cur][r * 48 + 16 + 8 * hf];  // K 16..23 / 24..31
        }
#pragma unroll
        for (int ni = 0; ni < 2; ni++) {
            const int c = wn * 32 + ni * 16 + l16;
            fb[ni].h[0] = *(const v8bf*)&Bs[cur][c * 48 + 8 * hf];       // K 0..7 / 8..15
            fb[ni].h[1] = *(const v8bf*)&Bs[cur][c * 48 + 16 + 8 * hf];  // K 16..23 / 24..31
        }
#pragma unroll
        for (int mi = 0; mi < 4; mi++)
#pragma unroll
            for (int ni = 0; ni < 2; ni++)
                acc[mi][ni] = __builtin_amdgcn_wmma_f32_16x16x32_bf16(
                    false, fa[mi].v, false, fb[ni].v, (short)0, acc[mi][ni], false, false);

        // 3) deposit next tile into alternate buffer
        if (more) {
            *(v8bf*)&As[nxt][ar * 48 + ac]     = na0;
            *(v8bf*)&As[nxt][ar * 48 + ac + 8] = na1;
#pragma unroll
            for (int i = 0; i < 8; i++) Bs[nxt][(bn0 + i) * 48 + bkr] = nb0.e[i];
#pragma unroll
            for (int i = 0; i < 8; i++) Bs[nxt][(bn0 + 8 + i) * 48 + bkr] = nb1.e[i];
        }
        __syncthreads();
    }

    // ---- epilogue: C/D layout: lane(0-15)->M=j, lane(16-31)->M=8+j, N=lane&15 ----
#pragma unroll
    for (int mi = 0; mi < 4; mi++)
#pragma unroll
        for (int ni = 0; ni < 2; ni++)
#pragma unroll
            for (int j = 0; j < 8; j++) {
                float v = acc[mi][ni][j];
                const long grow = rowBase + wm * 64 + mi * 16 + hf * 8 + j;
                const long gcol = colBase + wn * 32 + ni * 16 + l16;
                if constexpr (EPI == 0) {               // plain bf16 store (kv)
                    p.outb[(long)(z & 7) * p.ob_z1 + (long)(z >> 3) * p.ob_z8 +
                           grow * p.ob_ld + gcol] = (bf16)v;
                } else if constexpr (EPI == 1) {        // bias + rotary -> bf16 (q,k)
                    float t  = v + p.bias[gcol];
                    float pr = __shfl_xor(t, 1, 32);    // paired column lives in lane^1
                    const long l  = grow & (cL - 1);
                    const int  hd = (int)(gcol & (cD - 1));
                    const float s = p.aux[l * cD + (hd & ~1)];  // sin
                    const float c = p.aux[l * cD + (hd | 1)];   // cos
                    const float r = (hd & 1) ? (pr * s + t * c) : (t * c - pr * s);
                    p.outb[grow * p.ob_ld + gcol] = (bf16)r;
                } else if constexpr (EPI == 2) {        // bias -> bf16 (v)
                    p.outb[grow * p.ob_ld + gcol] = (bf16)(v + p.bias[gcol]);
                } else if constexpr (EPI == 3) {        // relu feature map -> f32 + bf16
                    const float y = fmaxf(v * 0.0625f, 0.0f) + 1e-6f;
                    p.outf[grow * cM + gcol] = y;
                    p.outb[grow * cM + gcol] = (bf16)y;
                } else if constexpr (EPI == 4) {        // attn / denom -> bf16 (head merge)
                    const long bidx = (long)(z >> 3);
                    const int  h    = z & 7;
                    const float dn  = p.aux[((bidx * cL + grow) << 3) + h];
                    p.outb[((bidx * cL + grow) << 10) + ((long)h << 7) + gcol] = (bf16)(v / dn);
                } else if constexpr (EPI == 5) {        // bias + residual -> f32 (Wo, W2)
                    p.outf[grow * p.of_ld + gcol] = v + p.bias[gcol] + p.aux[grow * p.aux_ld + gcol];
                } else if constexpr (EPI == 6) {        // bias + relu -> bf16 (W1)
                    p.outb[grow * p.ob_ld + gcol] = (bf16)fmaxf(v + p.bias[gcol], 0.0f);
                }
            }
}

// ---------------------------------------------------------------------------
__global__ LB void ln_kernel(const float* __restrict__ x, const float* __restrict__ g,
                             const float* __restrict__ b, bf16* __restrict__ out) {
    __shared__ float red[256];
    const int tid = threadIdx.x;
    const long row = blockIdx.x;
    const float* xr = x + row * cDM;
    float s = 0.f;
    for (int i = tid; i < cDM; i += 256) s += xr[i];
    red[tid] = s; __syncthreads();
    for (int o = 128; o > 0; o >>= 1) { if (tid < o) red[tid] += red[tid + o]; __syncthreads(); }
    const float mu = red[0] * (1.0f / cDM);
    __syncthreads();
    float vv = 0.f;
    for (int i = tid; i < cDM; i += 256) { float d = xr[i] - mu; vv += d * d; }
    red[tid] = vv; __syncthreads();
    for (int o = 128; o > 0; o >>= 1) { if (tid < o) red[tid] += red[tid + o]; __syncthreads(); }
    const float inv = rsqrtf(red[0] * (1.0f / cDM) + 1e-5f);
    for (int i = tid; i < cDM; i += 256)
        out[row * cDM + i] = (bf16)((xr[i] - mu) * inv * g[i] + b[i]);
}

__global__ LB void cvt_kernel(const float* __restrict__ s, bf16* __restrict__ d, long n) {
    long i = (long)blockIdx.x * 256 + threadIdx.x;
    const long st = (long)gridDim.x * 256;
    for (; i < n; i += st) d[i] = (bf16)s[i];
}

__global__ void projT_kernel(const float* __restrict__ proj, bf16* __restrict__ pt) {
    const int m = blockIdx.x;   // 0..255
    const int d = threadIdx.x;  // 0..127
    pt[d * cM + m] = (bf16)proj[m * cD + d];
}

__global__ void zero_kernel(float* p, int n) {
    const int i = blockIdx.x * 256 + threadIdx.x;
    if (i < n) p[i] = 0.f;
}

// kpT[((b*H+h)*M + m)*L + l] = kp[((b*L+l)*H + h)*M + m]   (tiled 32x32 via LDS)
__global__ LB void transpose_kernel(const bf16* __restrict__ kp, bf16* __restrict__ kpT) {
    __shared__ bf16 t[32][33];
    const int bh = blockIdx.z;
    const int b = bh >> 3, h = bh & 7;
    const int l0 = blockIdx.x * 32, m0 = blockIdx.y * 32;
    const int tx = threadIdx.x & 31, ty = threadIdx.x >> 5;  // 32 x 8
#pragma unroll
    for (int i = 0; i < 4; i++) {
        const int l = l0 + ty + 8 * i;
        t[ty + 8 * i][tx] = kp[((long)(b * cL + l) * cH + h) * cM + m0 + tx];
    }
    __syncthreads();
#pragma unroll
    for (int i = 0; i < 4; i++) {
        const int m = m0 + ty + 8 * i;
        kpT[((long)bh * cM + m) * cL + l0 + tx] = t[tx][ty + 8 * i];
    }
}

// ks[b,h,m] += sum over a 256-row chunk of l of k'[b,l,h,m]
__global__ LB void kssum_kernel(const bf16* __restrict__ kp, float* __restrict__ ks) {
    const int bh = blockIdx.x;
    const int b = bh >> 3, h = bh & 7;
    const int m = threadIdx.x;
    const long base = ((long)b * cL * cH + h) * (long)cM + m;
    const long l0 = (long)blockIdx.y * 256;
    float s = 0.f;
    for (long l = l0; l < l0 + 256; ++l) s += (float)kp[base + l * (cH * cM)];
    atomicAdd(&ks[(long)bh * cM + m], s);
}

// denom[(b*L+l)*H+h] = sum_m q'[b,l,h,m] * ks[b,h,m]   (one wave per output)
__global__ LB void denom_kernel(const bf16* __restrict__ qp, const float* __restrict__ ks,
                                float* __restrict__ dn) {
    const int lane = threadIdx.x & 31, wv = threadIdx.x >> 5;
    const long row = (long)blockIdx.x * 8 + wv;   // (b*L + l)*H + h
    const int b = (int)(row >> 15);               // L*H = 32768
    const int h = (int)(row & 7);
    const bf16* q = qp + row * cM;
    const float* k = ks + ((long)b * cH + h) * cM;
    float s = 0.f;
    for (int m = lane; m < cM; m += 32) s += (float)q[m] * k[m];
    for (int o = 16; o; o >>= 1) s += __shfl_xor(s, o, 32);
    if (lane == 0) dn[row] = s;
}

// ---------------------------------------------------------------------------
extern "C" void kernel_launch(void* const* d_in, const int* in_sizes, int n_in,
                              void* d_out, int out_size, void* d_ws, size_t ws_size,
                              hipStream_t stream) {
    (void)in_sizes; (void)n_in; (void)out_size; (void)ws_size;

    const float* inputs = (const float*)d_in[0];
    const float* rpe    = (const float*)d_in[1];
    const float* ln_g   = (const float*)d_in[2];
    const float* ln_b   = (const float*)d_in[3];
    const float* Wq = (const float*)d_in[4];  const float* bq = (const float*)d_in[5];
    const float* Wk = (const float*)d_in[6];  const float* bk = (const float*)d_in[7];
    const float* Wv = (const float*)d_in[8];  const float* bv = (const float*)d_in[9];
    const float* Wo = (const float*)d_in[10]; const float* bo = (const float*)d_in[11];
    const float* proj = (const float*)d_in[12];
    const float* fg = (const float*)d_in[13]; const float* fb = (const float*)d_in[14];
    const float* W1 = (const float*)d_in[15]; const float* b1 = (const float*)d_in[16];
    const float* W2 = (const float*)d_in[17]; const float* b2 = (const float*)d_in[18];

    const long NT = (long)cB * cL;                // 8192 tokens
    float* out_y  = (float*)d_out;                // (B,L,DM)
    float* out_kp = out_y + NT * cDM;             // (B,L,H,M)
    float* out_qp = out_kp + NT * cH * cM;        // (B,L,H,M)

    char* w = (char*)d_ws;
    size_t off = 0;
    auto alloc = [&](size_t bytes) -> void* {
        void* r = w + off;
        off = (off + bytes + 255) & ~(size_t)255;
        return r;
    };
    bf16* wq_h = (bf16*)alloc(1048576 * 2);
    bf16* wk_h = (bf16*)alloc(1048576 * 2);
    bf16* wv_h = (bf16*)alloc(1048576 * 2);
    bf16* wo_h = (bf16*)alloc(1048576 * 2);
    bf16* w1_h = (bf16*)alloc(2097152 * 2);
    bf16* w2_h = (bf16*)alloc(2097152 * 2);
    bf16* pt_h = (bf16*)alloc((size_t)cD * cM * 2);
    char* regA  = (char*)alloc(NT * cDM * 2);         // xh | attnh
    char* regB  = (char*)alloc(NT * cDM * 2);         // qrh | hln
    char* regCD = (char*)alloc(NT * cDM * 4);         // krh+vh | mha(f32)
    char* regE  = (char*)alloc(NT * cH * cM * 2);     // qp_h | h1
    char* regF  = (char*)alloc(NT * cH * cM * 2);     // kp_h
    bf16* kpT   = (bf16*)alloc((size_t)cB * cH * cM * cL * 2);
    bf16* kvh   = (bf16*)alloc((size_t)cB * cH * cM * cD * 2);
    float* ks    = (float*)alloc((size_t)cB * cH * cM * 4);
    float* denom = (float*)alloc((size_t)NT * cH * 4);

    bf16* xh    = (bf16*)regA;
    bf16* attnh = (bf16*)regA;
    bf16* qrh   = (bf16*)regB;
    bf16* hln   = (bf16*)regB;
    bf16* krh   = (bf16*)regCD;
    bf16* vh    = (bf16*)(regCD + NT * cDM * 2);
    float* mha  = (float*)regCD;
    bf16* qp_h  = (bf16*)regE;
    bf16* h1    = (bf16*)regE;
    bf16* kp_h  = (bf16*)regF;

    // weight conversions
    cvt_kernel<<<1024, 256, 0, stream>>>(Wq, wq_h, 1048576);
    cvt_kernel<<<1024, 256, 0, stream>>>(Wk, wk_h, 1048576);
    cvt_kernel<<<1024, 256, 0, stream>>>(Wv, wv_h, 1048576);
    cvt_kernel<<<1024, 256, 0, stream>>>(Wo, wo_h, 1048576);
    cvt_kernel<<<2048, 256, 0, stream>>>(W1, w1_h, 2097152);
    cvt_kernel<<<2048, 256, 0, stream>>>(W2, w2_h, 2097152);
    projT_kernel<<<cM, cD, 0, stream>>>(proj, pt_h);

    // LN1
    ln_kernel<<<(int)NT, 256, 0, stream>>>(inputs, ln_g, ln_b, xh);

    // Q = rotary(xh @ Wq + bq) -> bf16
    {
        GemmP g{}; g.A = xh; g.a_rs = cDM;
        g.Bm = wq_h; g.b_rs = cDM;
        g.outb = qrh; g.ob_ld = cDM; g.bias = bq; g.aux = rpe; g.K = cDM;
        gemm_bf16<1><<<dim3(cDM / 128, (int)(NT / 128), 1), 256, 0, stream>>>(g);
    }
    // K = rotary(xh @ Wk + bk)
    {
        GemmP g{}; g.A = xh; g.a_rs = cDM;
        g.Bm = wk_h; g.b_rs = cDM;
        g.outb = krh; g.ob_ld = cDM; g.bias = bk; g.aux = rpe; g.K = cDM;
        gemm_bf16<1><<<dim3(cDM / 128, (int)(NT / 128), 1), 256, 0, stream>>>(g);
    }
    // V = xh @ Wv + bv
    {
        GemmP g{}; g.A = xh; g.a_rs = cDM;
        g.Bm = wv_h; g.b_rs = cDM;
        g.outb = vh; g.ob_ld = cDM; g.bias = bv; g.K = cDM;
        gemm_bf16<2><<<dim3(cDM / 128, (int)(NT / 128), 1), 256, 0, stream>>>(g);
    }
    // q' = relu(q_rot @ projT / 16) + eps  -> out_qp (f32) + qp_h (bf16)
    {
        GemmP g{}; g.A = qrh; g.a_rs = cD;
        g.Bm = pt_h; g.b_rs = cM;
        g.outf = out_qp; g.outb = qp_h; g.K = cD;
        gemm_bf16<3><<<dim3(cM / 128, (int)(NT * cH / 128), 1), 256, 0, stream>>>(g);
    }
    // k'
    {
        GemmP g{}; g.A = krh; g.a_rs = cD;
        g.Bm = pt_h; g.b_rs = cM;
        g.outf = out_kp; g.outb = kp_h; g.K = cD;
        gemm_bf16<3><<<dim3(cM / 128, (int)(NT * cH / 128), 1), 256, 0, stream>>>(g);
    }
    // ks_sum and k' transpose (for unit-stride kv GEMM)
    zero_kernel<<<16, 256, 0, stream>>>(ks, cB * cH * cM);
    kssum_kernel<<<dim3(cB * cH, cL / 256), 256, 0, stream>>>(kp_h, ks);
    transpose_kernel<<<dim3(cL / 32, cM / 32, cB * cH), 256, 0, stream>>>(kp_h, kpT);
    // kv[b,h,m,d] = sum_l k'[b,l,h,m] v[b,l,h,d]   (A = kpT, unit l-stride)
    {
        GemmP g{}; g.A = kpT; g.a_z1 = (long)cM * cL; g.a_z8 = 8L * cM * cL; g.a_rs = cL;
        g.Bm = vh; g.b_z1 = cD; g.b_z8 = (long)cL * cH * cD; g.b_rs = cH * cD;
        g.outb = kvh; g.ob_z1 = cM * cD; g.ob_z8 = 8L * cM * cD; g.ob_ld = cD; g.K = cL;
        gemm_bf16<0><<<dim3(1, cM / 128, cB * cH), 256, 0, stream>>>(g);
    }
    // denom
    denom_kernel<<<(int)(NT * cH / 8), 256, 0, stream>>>(qp_h, ks, denom);
    // attn = (q' @ kv) / denom -> attnh (bf16, heads merged to DM)
    {
        GemmP g{}; g.A = qp_h; g.a_z1 = cM; g.a_z8 = (long)cL * cH * cM; g.a_rs = cH * cM;
        g.Bm = kvh; g.b_z1 = cM * cD; g.b_z8 = 8L * cM * cD; g.b_rs = cD;
        g.outb = attnh; g.aux = denom; g.K = cM;
        gemm_bf16<4><<<dim3(1, cL / 128, cB * cH), 256, 0, stream>>>(g);
    }
    // mha = attn @ Wo + bo + inputs  (f32)
    {
        GemmP g{}; g.A = attnh; g.a_rs = cDM;
        g.Bm = wo_h; g.b_rs = cDM;
        g.outf = mha; g.of_ld = cDM; g.bias = bo; g.aux = inputs; g.aux_ld = cDM; g.K = cDM;
        gemm_bf16<5><<<dim3(cDM / 128, (int)(NT / 128), 1), 256, 0, stream>>>(g);
    }
    // LN2
    ln_kernel<<<(int)NT, 256, 0, stream>>>(mha, fg, fb, hln);
    // h1 = relu(hln @ W1 + b1)
    {
        GemmP g{}; g.A = hln; g.a_rs = cDM;
        g.Bm = w1_h; g.b_rs = 2 * cDM;
        g.outb = h1; g.ob_ld = 2 * cDM; g.bias = b1; g.K = cDM;
        gemm_bf16<6><<<dim3(2 * cDM / 128, (int)(NT / 128), 1), 256, 0, stream>>>(g);
    }
    // y = h1 @ W2 + b2 + mha  (f32 -> d_out)
    {
        GemmP g{}; g.A = h1; g.a_rs = 2 * cDM;
        g.Bm = w2_h; g.b_rs = cDM;
        g.outf = out_y; g.of_ld = cDM; g.bias = b2; g.aux = mha; g.aux_ld = cDM; g.K = 2 * cDM;
        gemm_bf16<5><<<dim3(cDM / 128, (int)(NT / 128), 1), 256, 0, stream>>>(g);
    }
}
